// CloudCrop_1821066134100
// MI455X (gfx1250) — matrix operand order repaired and checked by reading, not compiled, not run on hardware
//
#include <hip/hip_runtime.h>
#include <hip/hip_bf16.h>

typedef __attribute__((ext_vector_type(16))) _Float16 v16h;
typedef __attribute__((ext_vector_type(8)))  _Float16 h8;
typedef __attribute__((ext_vector_type(8)))  float    v8f;

#define B_    2
#define S_    1024
#define N_    20000
#define D_    4
#define NS_   64
#define RAD2_ 0.0025f
#define HMIN_ (-0.02f)
#define EPS_  1e-5f

// ---------------------------------------------------------------------------
// Kernel 0: fold BN into scale/bias, convert weights to f16 (W1 K-padded 3->32)
// ---------------------------------------------------------------------------
__global__ __launch_bounds__(256) void prep_kernel(
    const float* __restrict__ w1, const float* __restrict__ w2, const float* __restrict__ w3,
    const float* __restrict__ g1, const float* __restrict__ be1, const float* __restrict__ mn1, const float* __restrict__ vr1,
    const float* __restrict__ g2, const float* __restrict__ be2, const float* __restrict__ mn2, const float* __restrict__ vr2,
    const float* __restrict__ g3, const float* __restrict__ be3, const float* __restrict__ mn3, const float* __restrict__ vr3,
    _Float16* __restrict__ W1h, _Float16* __restrict__ W2h, _Float16* __restrict__ W3h,
    float* __restrict__ SB)
{
    const int t   = blockIdx.x * blockDim.x + threadIdx.x;
    const int nth = gridDim.x * blockDim.x;
    for (int i = t; i < 64 * 32; i += nth) {
        int o = i >> 5, c = i & 31;
        W1h[i] = (_Float16)(c < 3 ? w1[o * 3 + c] : 0.f);
    }
    for (int i = t; i < 128 * 64;  i += nth) W2h[i] = (_Float16)w2[i];
    for (int i = t; i < 256 * 128; i += nth) W3h[i] = (_Float16)w3[i];
    for (int i = t; i < 448; i += nth) {
        float g, be, mn, vr;
        if (i < 64)       { g = g1[i];       be = be1[i];       mn = mn1[i];       vr = vr1[i]; }
        else if (i < 192) { int j = i - 64;  g = g2[j]; be = be2[j]; mn = mn2[j]; vr = vr2[j]; }
        else              { int j = i - 192; g = g3[j]; be = be3[j]; mn = mn3[j]; vr = vr3[j]; }
        float sc = g * rsqrtf(vr + EPS_);
        SB[i]       = sc;
        SB[448 + i] = be - mn * sc;
    }
}

// ---------------------------------------------------------------------------
// Kernel 1: cylinder grouping. One wave32 per seed; in-order ballot append
// picks the first 64 passing point indices per hmax bin (matches top_k trick).
// ---------------------------------------------------------------------------
__global__ __launch_bounds__(32) void group_kernel(
    const float* __restrict__ seed, const float* __restrict__ pc,
    const float* __restrict__ rot,  float* __restrict__ Xg)
{
    const int bs   = blockIdx.x;           // b*S + s
    const int lane = threadIdx.x;
    const int b    = bs / S_;
    const float sx = seed[bs * 3 + 0], sy = seed[bs * 3 + 1], sz = seed[bs * 3 + 2];
    float R[9];
#pragma unroll
    for (int i = 0; i < 9; i++) R[i] = rot[bs * 9 + i];
    const float* pcb = pc + (size_t)b * N_ * 3;
    const float hmax[4] = {0.01f, 0.02f, 0.03f, 0.04f};
    int   cnt[4] = {0, 0, 0, 0};
    float f0x[4] = {0, 0, 0, 0}, f0y[4] = {0, 0, 0, 0}, f0z[4] = {0, 0, 0, 0};
    float p0x = 0.f, p0y = 0.f, p0z = 0.f;
    const unsigned ltmask = (1u << lane) - 1u;

    for (int base = 0; base < N_; base += 32) {          // N_ % 32 == 0
        const int i = base + lane;
        float px = pcb[i * 3 + 0], py = pcb[i * 3 + 1], pz = pcb[i * 3 + 2];
        if (base + 32 < N_) __builtin_prefetch(pcb + (i + 32) * 3, 0, 1);
        float ex = px - sx, ey = py - sy, ez = pz - sz;
        // rel_rot[d] = sum_c rel[c] * R[c][d]
        float rx = ex * R[0] + ey * R[3] + ez * R[6];
        float ry = ex * R[1] + ey * R[4] + ez * R[7];
        float rz = ex * R[2] + ey * R[5] + ez * R[8];
        if (base == 0) { p0x = __shfl(rx, 0, 32); p0y = __shfl(ry, 0, 32); p0z = __shfl(rz, 0, 32); }
        float r2  = ry * ry + rz * rz;
        bool  inC = (r2 < RAD2_) && (rx > HMIN_);
#pragma unroll
        for (int d = 0; d < 4; d++) {
            bool m = inC && (rx < hmax[d]);
            unsigned mask = (unsigned)__ballot(m);
            if (cnt[d] == 0 && mask) {                   // remember first passing value
                int src = __ffs(mask) - 1;
                f0x[d] = __shfl(rx, src, 32);
                f0y[d] = __shfl(ry, src, 32);
                f0z[d] = __shfl(rz, src, 32);
            }
            if (cnt[d] < NS_) {
                int pos = cnt[d] + __popc(mask & ltmask);
                if (m && pos < NS_) {
                    float* xp = Xg + (((size_t)bs * 4 + d) * NS_ + pos) * 3;
                    xp[0] = rx; xp[1] = ry; xp[2] = rz;
                }
                cnt[d] += __popc(mask);
                if (cnt[d] > NS_) cnt[d] = NS_;
            }
        }
        if (cnt[0] == NS_ && cnt[1] == NS_ && cnt[2] == NS_ && cnt[3] == NS_) break;
    }
    // pad remaining slots with slot-0 value (or point 0's rel_rot if no hits)
#pragma unroll
    for (int d = 0; d < 4; d++) {
        float fx, fy, fz;
        if (cnt[d] > 0) { fx = f0x[d]; fy = f0y[d]; fz = f0z[d]; }
        else            { fx = p0x;    fy = p0y;    fz = p0z;    }
        for (int n = cnt[d] + lane; n < NS_; n += 32) {
            float* xp = Xg + (((size_t)bs * 4 + d) * NS_ + n) * 3;
            xp[0] = fx; xp[1] = fy; xp[2] = fz;
        }
    }
}

// ---------------------------------------------------------------------------
// WMMA fragment helpers (wave32, v_wmma_f32_16x16x32_f16 A and B both in contiguous k order k = 16*(L/16)+e)
// ---------------------------------------------------------------------------
__device__ __forceinline__ v8f wmma32(v16h a, v16h b, v8f c) {
    return __builtin_amdgcn_wmma_f32_16x16x32_f16(false, a, false, b, (short)0, c, false, false);
}

// A (MxK, f16, row-major, K-contiguous): lane L -> M = mbase + L%16;
// elems e = K (L<16?0:16)+e (0..15)
__device__ __forceinline__ v16h load_a_frag(const _Float16* __restrict__ W, int C,
                                            int mbase, int kbase, int lane) {
    int m    = mbase + (lane & 15);
    int koff = (lane < 16) ? 0 : 16;
    const _Float16* p = W + (size_t)m * C + kbase + koff;
    h8 lo = *(const h8*)p;
    h8 hi = *(const h8*)(p + 8);
    v16h r;
#pragma unroll
    for (int i = 0; i < 8; i++) { r[i] = lo[i]; r[i + 8] = hi[i]; }
    return r;
}

// B (KxN, f16) staged in LDS column-major [col][C]: lane L -> N = colbase + L%16;
// elems e -> K = kbase + (L<16?0:16) + e  (16 contiguous halves, 32B-aligned)
__device__ __forceinline__ v16h load_b_frag(const _Float16* H, int C,
                                            int colbase, int kbase, int lane) {
    int col  = colbase + (lane & 15);
    int koff = (lane < 16) ? 0 : 16;
    return *(const v16h*)(H + (size_t)col * C + kbase + koff);
}

// D frag (16x16 f32): VGPR r -> M = mbase + r + ((lane&16)?8:0), col = colbase + lane%16
__device__ __forceinline__ void bn_relu_store(v8f dacc, const float* sS, const float* sB,
                                              int mbase, int colbase,
                                              _Float16* H, int C, int lane) {
    int col  = colbase + (lane & 15);
    int madd = (lane & 16) ? 8 : 0;
#pragma unroll
    for (int r = 0; r < 8; r++) {
        int   m = mbase + r + madd;
        float v = dacc[r] * sS[m] + sB[m];
        v = v > 0.f ? v : 0.f;
        H[(size_t)col * C + m] = (_Float16)v;
    }
}

// ---------------------------------------------------------------------------
// Kernel 2: shared MLP (3->64->128->256, BN+ReLU) + max over 64 samples.
// One block (4 waves) per (b,s,d) group of 64 columns. 336 WMMA per block.
// Weight A-fragments hoisted to registers (loaded once per (mi,kt), not per nt).
// ---------------------------------------------------------------------------
__global__ __launch_bounds__(128) void mlp_kernel(
    const float* __restrict__ Xg,
    const _Float16* __restrict__ W1h, const _Float16* __restrict__ W2h,
    const _Float16* __restrict__ W3h, const float* __restrict__ SBg,
    float* __restrict__ out)
{
    __shared__ __align__(32) _Float16 lX [64 * 32];
    __shared__ __align__(32) _Float16 lH1[64 * 64];
    __shared__ __align__(32) _Float16 lH2[64 * 128];
    __shared__ float sS[448];
    __shared__ float sB[448];

    const int q    = blockIdx.x;          // (b*S + s)*4 + d
    const int tid  = threadIdx.x;
    const int lane = tid & 31;
    const int w    = tid >> 5;            // wave 0..3
    const int b    = q >> 12;
    const int sd   = q & 4095;
    const int s    = sd >> 2;
    const int d    = sd & 3;

    // Stage input columns (f32 -> f16, K padded to 32) and BN params
    const float* xcol = Xg + (size_t)q * (NS_ * 3);
    for (int i = tid; i < 64 * 32; i += 128) {
        int col = i >> 5, k = i & 31;
        lX[i] = (_Float16)(k < 3 ? xcol[col * 3 + k] : 0.f);
    }
    for (int i = tid; i < 448; i += 128) { sS[i] = SBg[i]; sB[i] = SBg[448 + i]; }
    __syncthreads();

    // -------- Layer 1: 64x32 @ 32x64 (wave w owns M-tile w) --------
    {
        const int mbase = w * 16;
        const v16h a = load_a_frag(W1h, 32, mbase, 0, lane);   // hoisted: 1 load
#pragma unroll
        for (int nt = 0; nt < 4; nt++) {
            v16h bf = load_b_frag(lX, 32, nt * 16, 0, lane);
            v8f acc = {};
            acc = wmma32(a, bf, acc);
            bn_relu_store(acc, sS + 0, sB + 0, mbase, nt * 16, lH1, 64, lane);
        }
    }
    __syncthreads();

    // -------- Layer 2: 128x64 @ 64x64 (wave w owns M-tiles w, w+4) --------
#pragma unroll
    for (int mi = 0; mi < 2; mi++) {
        const int mbase = (w + mi * 4) * 16;
        const v16h a0 = load_a_frag(W2h, 64, mbase, 0,  lane); // hoisted: 2 loads
        const v16h a1 = load_a_frag(W2h, 64, mbase, 32, lane);
#pragma unroll
        for (int nt = 0; nt < 4; nt++) {
            v8f acc = {};
            acc = wmma32(a0, load_b_frag(lH1, 64, nt * 16, 0,  lane), acc);
            acc = wmma32(a1, load_b_frag(lH1, 64, nt * 16, 32, lane), acc);
            bn_relu_store(acc, sS + 64, sB + 64, mbase, nt * 16, lH2, 128, lane);
        }
    }
    __syncthreads();

    // -------- Layer 3: 256x128 @ 128x64, fused max over 64 cols --------
#pragma unroll
    for (int mi = 0; mi < 4; mi++) {
        const int mbase = (w * 4 + mi) * 16;
        v16h a[4];                                             // hoisted: 4 loads
#pragma unroll
        for (int kt = 0; kt < 4; kt++) a[kt] = load_a_frag(W3h, 128, mbase, kt * 32, lane);
        v8f mx = {};                       // ReLU output >= 0, so 0 is identity
#pragma unroll
        for (int nt = 0; nt < 4; nt++) {
            v8f acc = {};
#pragma unroll
            for (int kt = 0; kt < 4; kt++) {
                v16h bf = load_b_frag(lH2, 128, nt * 16, kt * 32, lane);
                acc = wmma32(a[kt], bf, acc);
            }
            const int madd = (lane & 16) ? 8 : 0;
#pragma unroll
            for (int r = 0; r < 8; r++) {
                int   m = mbase + r + madd;
                float v = acc[r] * sS[192 + m] + sB[192 + m];
                v = v > 0.f ? v : 0.f;
                if (v > mx[r]) mx[r] = v;
            }
        }
        // reduce max across the 16 lane-columns inside each half-wave
#pragma unroll
        for (int off = 1; off < 16; off <<= 1) {
#pragma unroll
            for (int r = 0; r < 8; r++) {
                float o = __shfl_xor(mx[r], off, 32);
                if (o > mx[r]) mx[r] = o;
            }
        }
        if ((lane & 15) == 0) {            // lanes 0 (M=r) and 16 (M=r+8) write
            const int madd = (lane & 16) ? 8 : 0;
#pragma unroll
            for (int r = 0; r < 8; r++) {
                int m = mbase + r + madd;
                out[(((size_t)b * 256 + m) * S_ + s) * D_ + d] = mx[r];
            }
        }
    }
}

// ---------------------------------------------------------------------------
extern "C" void kernel_launch(void* const* d_in, const int* in_sizes, int n_in,
                              void* d_out, int out_size, void* d_ws, size_t ws_size,
                              hipStream_t stream) {
    const float* seed = (const float*)d_in[0];
    const float* pc   = (const float*)d_in[1];
    const float* rot  = (const float*)d_in[2];
    const float* w1   = (const float*)d_in[3];
    const float* g1   = (const float*)d_in[4];
    const float* be1  = (const float*)d_in[5];
    const float* mn1  = (const float*)d_in[6];
    const float* vr1  = (const float*)d_in[7];
    const float* w2   = (const float*)d_in[8];
    const float* g2   = (const float*)d_in[9];
    const float* be2  = (const float*)d_in[10];
    const float* mn2  = (const float*)d_in[11];
    const float* vr2  = (const float*)d_in[12];
    const float* w3   = (const float*)d_in[13];
    const float* g3   = (const float*)d_in[14];
    const float* be3  = (const float*)d_in[15];
    const float* mn3  = (const float*)d_in[16];
    const float* vr3  = (const float*)d_in[17];

    char* ws = (char*)d_ws;
    float*    Xg  = (float*)ws;                       // 8192*64*3 f32 = 6,291,456 B
    _Float16* W1h = (_Float16*)(ws + 6291456);        //  4,096 B
    _Float16* W2h = (_Float16*)(ws + 6295552);        // 16,384 B
    _Float16* W3h = (_Float16*)(ws + 6311936);        // 65,536 B
    float*    SBf = (float*)(ws + 6377472);           //  3,584 B

    prep_kernel<<<64, 256, 0, stream>>>(w1, w2, w3,
                                        g1, be1, mn1, vr1,
                                        g2, be2, mn2, vr2,
                                        g3, be3, mn3, vr3,
                                        W1h, W2h, W3h, SBf);
    group_kernel<<<B_ * S_, 32, 0, stream>>>(seed, pc, rot, Xg);
    mlp_kernel<<<B_ * S_ * D_, 128, 0, stream>>>(Xg, W1h, W2h, W3h, SBf, (float*)d_out);
}
